// LSTM_80891414053414
// MI455X (gfx1250) — matrix operand order repaired and checked by reading, not compiled, not run on hardware
//
#include <hip/hip_runtime.h>
#include <hip/hip_bf16.h>
#include <math.h>

// ---- WMMA vector types (CDNA5 / gfx1250, wave32) ----
typedef __attribute__((ext_vector_type(16))) __bf16 v16bf;
typedef __attribute__((ext_vector_type(8)))  __bf16 v8bf;
typedef __attribute__((ext_vector_type(8)))  float  v8f;

#define N_BATCH 64
#define T_SEQ   512
#define D_IN    1024
#define H_DIM   1024
#define G4H     4096              // 4*H
#define KT      32                // K-tiles of 32 (1024/32)
#define FRAG_H  512               // halves per packed B fragment (32 lanes * 16)

// ------------------------------------------------------------------
// Phase 0a: f32 -> bf16 convert (for x)
// ------------------------------------------------------------------
__global__ void k_cvt_bf16(const float* __restrict__ src, __bf16* __restrict__ dst, int n) {
    int i = blockIdx.x * blockDim.x + threadIdx.x;
    if (i < n) dst[i] = (__bf16)src[i];
}

// ------------------------------------------------------------------
// Phase 0b: pack [1024 x 4096] f32 weight into WMMA B-fragment order.
// Fragment (ntile, ktile) covers B[k=ktile*32..+32][n=ntile*16..+16].
// 16x16x32 bf16 B layout (wave32): lanes 0-15 hold K=0..7,16..23,
// lanes 16-31 hold K=8..15,24..31 (elem e -> K=8h+e, e>=8 -> K=16+8h+e-8); N = lane&15.
// Each lane's 16 bf16 (32B) contiguous -> 2x global_load_b128 per fragment.
// frag_id = ntile*KT + ktile.
// ------------------------------------------------------------------
__global__ void k_pack_w(const float* __restrict__ w, __bf16* __restrict__ wp) {
    int id = blockIdx.x * blockDim.x + threadIdx.x;   // 0 .. 1024*4096-1
    int e     = id & 15;
    int lane  = (id >> 4) & 31;
    int frag  = id >> 9;
    int ktile = frag & (KT - 1);
    int ntile = frag >> 5;
    int klocal = ((lane < 16) ? 0 : 8) + ((e < 8) ? e : (e + 8));
    int k = ktile * 32 + klocal;
    int n = ntile * 16 + (lane & 15);
    wp[id] = (__bf16)w[(size_t)k * G4H + n];
}

// ------------------------------------------------------------------
// Phase 0c: h0 -> bf16, c0 = 0
// ------------------------------------------------------------------
__global__ void k_init_state(const float* __restrict__ h0, __bf16* __restrict__ hbf,
                             float* __restrict__ c) {
    int i = blockIdx.x * blockDim.x + threadIdx.x;
    if (i < N_BATCH * H_DIM) { hbf[i] = (__bf16)h0[i]; c[i] = 0.0f; }
}

// ------------------------------------------------------------------
// Phase 1: xW[T,N,4H] = x[N*T,D] @ Wx[D,4H] + b  (bf16 WMMA, f32 acc)
// 1 wave -> 16x16 M-tile x 4 N-tiles (A-fragment reuse), double-buffered
// fragment loads so WMMAs overlap the next k-tile's global loads.
// Output row permuted r = n*T+t -> orow = t*N+n ([T,N,4H] slab per step).
// ------------------------------------------------------------------
__global__ __launch_bounds__(256) void k_xw_gemm(
    const __bf16* __restrict__ xbf, const __bf16* __restrict__ wxp,
    const float* __restrict__ bias, float* __restrict__ xw) {
    const int lane = threadIdx.x & 31;
    const int job  = blockIdx.x * 8 + (threadIdx.x >> 5);
    const int mtile = job >> 6;     // 0..2047
    const int ngrp  = job & 63;     // 4 ntiles each

    const int arow = mtile * 16 + (lane & 15);
    const __bf16* arp = xbf + (size_t)arow * D_IN + ((lane < 16) ? 0 : 8);
    // B fragment (g, kt) at bp + g*(KT*FRAG_H) + kt*FRAG_H
    const __bf16* bp = wxp + (size_t)(ngrp * 4) * KT * FRAG_H + (size_t)lane * 16;

    v8f acc[4] = {};
    v16bf aF[2];
    v16bf bF[2][4];

    auto loadFrags = [&](int kt, int buf) {
        *((v8bf*)&aF[buf])     = *(const v8bf*)(arp + kt * 32);
        *((v8bf*)&aF[buf] + 1) = *(const v8bf*)(arp + kt * 32 + 16);
        #pragma unroll
        for (int g = 0; g < 4; ++g)
            bF[buf][g] = *(const v16bf*)(bp + (size_t)g * (KT * FRAG_H) + kt * FRAG_H);
    };

    loadFrags(0, 0);
    #pragma unroll 2
    for (int kt = 0; kt < KT; ++kt) {
        const int cur = kt & 1;
        if (kt + 1 < KT) loadFrags(kt + 1, cur ^ 1);   // prefetch next k-tile
        #pragma unroll
        for (int g = 0; g < 4; ++g) {
            acc[g] = __builtin_amdgcn_wmma_f32_16x16x32_bf16(
                false, aF[cur], false, bF[cur][g], (short)0, acc[g], false, false);
        }
    }

    const int mlo = (lane >> 4) << 3;      // C/D layout: lanes 16-31 hold M+8
    #pragma unroll
    for (int g = 0; g < 4; ++g) {
        const int col = (ngrp * 4 + g) * 16 + (lane & 15);
        const float bv = bias[col];
        #pragma unroll
        for (int e = 0; e < 8; ++e) {
            const int r = mtile * 16 + mlo + e;                 // r = n*T + t
            const int orow = (r & (T_SEQ - 1)) * N_BATCH + (r >> 9);
            xw[(size_t)orow * G4H + col] = acc[g][e] + bv;
        }
    }
}

// ------------------------------------------------------------------
// Phase 2: one fused LSTM step, K-split x4 with LDS reduction.
// Block = 4 waves, one (rowTile, colTile): wave w computes k-tiles
// [w*8, w*8+8) for all 4 gate tiles (A-fragment reuse), partial sums
// reduced through LDS, then 128 threads apply gating + c/h update.
// Grid (4, 64) -> 1024 waves per step.
// ------------------------------------------------------------------
__global__ __launch_bounds__(128) void k_lstm_step(
    const __bf16* __restrict__ hin, __bf16* __restrict__ hout,
    const __bf16* __restrict__ whp, const float* __restrict__ xw_t,
    float* __restrict__ c, float* __restrict__ out, int t) {
    __shared__ float red[4][4][256];             // [wave][gate][M*16+N], 16 KB

    const int lane = threadIdx.x & 31;
    const int wave = threadIdx.x >> 5;
    const int rowTile = blockIdx.x;              // 0..3   (batch tiles)
    const int colTile = blockIdx.y;              // 0..63  (H feature tiles)
    const int kt0 = wave * 8;                    // this wave's k-tile range

    // Prefetch the xW_t entries this thread will consume after the GEMM
    // (streamed once per step; hide behind the 32 WMMAs).
    {
        const int p0 = threadIdx.x;
        #pragma unroll
        for (int h = 0; h < 2; ++h) {
            const int p = p0 + h * 128;
            const int n = rowTile * 16 + (p >> 4);
            const int j = colTile * 16 + (p & 15);
            const float* ar = xw_t + (size_t)n * G4H + j;
            __builtin_prefetch(ar, 0, 1);
            __builtin_prefetch(ar + H_DIM, 0, 1);
            __builtin_prefetch(ar + 2 * H_DIM, 0, 1);
            __builtin_prefetch(ar + 3 * H_DIM, 0, 1);
        }
    }

    const int arow = rowTile * 16 + (lane & 15);
    const __bf16* arp = hin + (size_t)arow * H_DIM + ((lane < 16) ? 0 : 8);
    // ntile = gi*64 + colTile; fragment (gi, kt) at
    // bp + gi*(64*KT*FRAG_H) + kt*FRAG_H
    const __bf16* bp = whp + (size_t)colTile * (KT * FRAG_H) + (size_t)lane * 16;

    v8f acc[4] = {};
    v16bf aF[2];
    v16bf bF[2][4];

    auto loadFrags = [&](int kt, int buf) {
        *((v8bf*)&aF[buf])     = *(const v8bf*)(arp + kt * 32);
        *((v8bf*)&aF[buf] + 1) = *(const v8bf*)(arp + kt * 32 + 16);
        #pragma unroll
        for (int gi = 0; gi < 4; ++gi)
            bF[buf][gi] = *(const v16bf*)(bp + (size_t)gi * (64 * KT * FRAG_H)
                                             + (size_t)kt * FRAG_H);
    };

    loadFrags(kt0, 0);
    #pragma unroll
    for (int k = 0; k < 8; ++k) {
        const int cur = k & 1;
        if (k + 1 < 8) loadFrags(kt0 + k + 1, cur ^ 1);
        #pragma unroll
        for (int gi = 0; gi < 4; ++gi) {
            acc[gi] = __builtin_amdgcn_wmma_f32_16x16x32_bf16(
                false, aF[cur], false, bF[cur][gi], (short)0, acc[gi], false, false);
        }
    }

    // Partials -> LDS
    const int mlo = (lane >> 4) << 3;
    #pragma unroll
    for (int gi = 0; gi < 4; ++gi) {
        #pragma unroll
        for (int e = 0; e < 8; ++e)
            red[wave][gi][(mlo + e) * 16 + (lane & 15)] = acc[gi][e];
    }
    __syncthreads();

    // Reduce 4 partials and apply the fused LSTM gating; 2 elems/thread.
    #pragma unroll
    for (int h = 0; h < 2; ++h) {
        const int p = threadIdx.x + h * 128;     // p = M*16 + N
        const int n = rowTile * 16 + (p >> 4);
        const int j = colTile * 16 + (p & 15);
        const float* ar = xw_t + (size_t)n * G4H + j;
        const float ai = red[0][0][p] + red[1][0][p] + red[2][0][p] + red[3][0][p] + ar[0];
        const float af = red[0][1][p] + red[1][1][p] + red[2][1][p] + red[3][1][p] + ar[H_DIM];
        const float ao = red[0][2][p] + red[1][2][p] + red[2][2][p] + red[3][2][p] + ar[2 * H_DIM];
        const float ag = red[0][3][p] + red[1][3][p] + red[2][3][p] + red[3][3][p] + ar[3 * H_DIM];
        const float ig = 1.0f / (1.0f + expf(-ai));
        const float fg = 1.0f / (1.0f + expf(-af));
        const float og = 1.0f / (1.0f + expf(-ao));
        const float gg = tanhf(ag);
        const int cidx = n * H_DIM + j;
        const float cv = fg * c[cidx] + ig * gg;
        c[cidx] = cv;
        const float hv = og * tanhf(cv);
        hout[cidx] = (__bf16)hv;
        out[((size_t)n * T_SEQ + t) * H_DIM + j] = hv;
    }
}

// ------------------------------------------------------------------
// Host side: phase 0 converts/packs, phase 1 big GEMM, phase 2 = 512
// sequential fused step kernels (ping-pong h buffers). All on `stream`,
// graph-capture safe (kernel launches only).
// ------------------------------------------------------------------
extern "C" void kernel_launch(void* const* d_in, const int* in_sizes, int n_in,
                              void* d_out, int out_size, void* d_ws, size_t ws_size,
                              hipStream_t stream) {
    const float* x  = (const float*)d_in[0];   // [N,T,D]
    const float* h0 = (const float*)d_in[1];   // [N,H]
    const float* Wx = (const float*)d_in[2];   // [D,4H]
    const float* Wh = (const float*)d_in[3];   // [H,4H]
    const float* b  = (const float*)d_in[4];   // [4H]
    float* out = (float*)d_out;                // [N,T,H]

    char* ws = (char*)d_ws;
    __bf16* xbf = (__bf16*)(ws);                    //  64 MB  x bf16
    __bf16* wxp = (__bf16*)(ws +  67108864);        // 8.0 MB  Wx packed
    __bf16* whp = (__bf16*)(ws +  75497472);        // 8.0 MB  Wh packed
    float*  xw  = (float*) (ws +  83886080);        // 512 MB  xW f32 [T,N,4H]
    __bf16* hA  = (__bf16*)(ws + 620756992);        // 128 KB  h ping
    __bf16* hB  = (__bf16*)(ws + 620888064);        // 128 KB  h pong
    float*  c   = (float*) (ws + 621019136);        // 256 KB  cell state

    // Phase 0
    k_cvt_bf16<<<(N_BATCH * T_SEQ * D_IN) / 256, 256, 0, stream>>>(
        x, xbf, N_BATCH * T_SEQ * D_IN);
    k_pack_w<<<(D_IN * G4H) / 256, 256, 0, stream>>>(Wx, wxp);
    k_pack_w<<<(H_DIM * G4H) / 256, 256, 0, stream>>>(Wh, whp);
    k_init_state<<<(N_BATCH * H_DIM) / 256, 256, 0, stream>>>(h0, hA, c);

    // Phase 1: 2048 mtiles * 64 ngroups = 131072 wave-jobs, 8 waves/block
    k_xw_gemm<<<16384, 256, 0, stream>>>(xbf, wxp, b, xw);

    // Phase 2: 512 sequential fused steps
    for (int t = 0; t < T_SEQ; ++t) {
        const __bf16* hin = (t & 1) ? hB : hA;
        __bf16*      hout = (t & 1) ? hA : hB;
        k_lstm_step<<<dim3(4, 64), 128, 0, stream>>>(
            hin, hout, whp, xw + (size_t)t * N_BATCH * G4H, c, out, t);
    }
}
